// Attention_53876069761431
// MI455X (gfx1250) — compile-verified, matrix operand order checked by reading, not run on hardware
//
#include <hip/hip_runtime.h>
#include <hip/hip_bf16.h>
#include <math.h>

// MI455X / gfx1250, wave32.
// GEMM [32*4096,512]@[512,512] via v_wmma_f32_16x16x32_f16 (f32 accum), fused
// with tanh + context-vector reduction so h is never materialized.
// 64-row M-tiles; WMMAs grouped by A-fragment so all four B fragments stay
// live -> 8-load B clause with laddered s_wait_loadcnt instead of four
// serialized wait-0s per K-step (the short-latency LDS A loads are the
// serialized resource instead of the L2 B loads).

typedef _Float16 v16h __attribute__((ext_vector_type(16)));
typedef float    v8f  __attribute__((ext_vector_type(8)));

#define B_DIM  32
#define T_DIM  4096
#define HID2   512
#define H_CLS  1024
#define D_OUT  8
#define MROWS  64            // T-rows per block in the GEMM phase
#define NCHUNK 16            // T-chunks for the pooling pass
#define CH_T   (T_DIM / NCHUNK)

// 4-op tanh: 1 - 2/(exp(2x)+1) using the hardware v_exp transcendental.
__device__ __forceinline__ float fast_tanh(float x) {
    return 1.0f - 2.0f / (__expf(2.0f * x) + 1.0f);
}

// K index inside a 32-wide K step for 16-bit A/B fragments (ISA 7.12.2):
// lanes 0-15 hold K octets {0-7,16-23}, lanes 16-31 hold {8-15,24-31};
// element pairs (2i,2i+1) are K-consecutive.
__device__ __forceinline__ int kin_map(int e, int khalf) {
    return khalf * 8 + e + ((e >= 8) ? 8 : 0);
}

// ---------------------------------------------------------------------------
// Kernel 0: weight [512,512] f32 (k-major) -> f16 pre-swizzled B fragments:
// wf[((nt*16 + ks)*32 + lane)*16 + e]. One element per thread.
// ---------------------------------------------------------------------------
__global__ __launch_bounds__(256)
void swizzle_weight_f16(const float* __restrict__ w, _Float16* __restrict__ wf) {
    int idx   = blockIdx.x * blockDim.x + threadIdx.x;   // 0 .. 512*512-1
    int e     = idx & 15;
    int lane  = (idx >> 4) & 31;
    int ks    = (idx >> 9) & 15;
    int nt    = idx >> 13;
    int n     = nt * 16 + (lane & 15);
    int khalf = lane >> 4;
    int k     = ks * 32 + kin_map(e, khalf);
    wf[idx] = (_Float16)w[k * HID2 + n];
}

// ---------------------------------------------------------------------------
// Kernel 1: scores[b,t] = tanh( sum_n tanh((emb_t . W_n) + bias_n) * cw_n )
// Block = 256 threads (8 waves) owns 64 T-rows of one batch.
// Wave: 4 N-tiles x 4 M-subtiles; per K-step: 4 B + 4 A frags -> 16 WMMAs.
// ---------------------------------------------------------------------------
__global__ __launch_bounds__(256)
void scores_kernel(const float* __restrict__ emb, const _Float16* __restrict__ wf,
                   const float* __restrict__ bias, const float* __restrict__ cw,
                   float* __restrict__ scores) {
    // 64 KB A-fragment stage; reused as the cross-wave score reduction
    // buffer after the K loop (A frags are dead by then).
    __shared__ __align__(64) _Float16 aswz[4 * 16 * 32 * 16];

    const int b    = blockIdx.x >> 6;              // 64 T-tiles per batch
    const int t0   = (blockIdx.x & 63) * MROWS;
    const int tid  = threadIdx.x;
    const int lane = tid & 31;
    const int wave = tid >> 5;

    // Stage A tile (64 rows x 512 K) as f16 fragments in LDS.
    // chunk c = msub*512 + ks*32 + frag_lane ; 16 halves per chunk.
    for (int c = tid; c < 4 * 16 * 32; c += 256) {
        int clane = c & 31;
        int ks    = (c >> 5) & 15;
        int msub  = c >> 9;
        int row   = t0 + msub * 16 + (clane & 15);
        int khalf = clane >> 4;
        const float* src = emb + ((size_t)b * T_DIM + row) * HID2 + ks * 32;
        _Float16* dst = &aswz[c * 16];
#pragma unroll
        for (int p = 0; p < 8; ++p) {          // element pairs are K-consecutive
            float2 v = *(const float2*)&src[kin_map(2 * p, khalf)];
            dst[2 * p]     = (_Float16)v.x;
            dst[2 * p + 1] = (_Float16)v.y;
        }
    }
    __syncthreads();

    // Accumulators: acc[nti][msub], 128 VGPRs.
    v8f acc[4][4];
#pragma unroll
    for (int i = 0; i < 4; ++i)
#pragma unroll
        for (int m = 0; m < 4; ++m)
            acc[i][m] = (v8f){0.f,0.f,0.f,0.f,0.f,0.f,0.f,0.f};

    // One base pointer; every B fragment is an immediate offset from it.
    const _Float16* wbase = wf + ((size_t)wave * 32768 + (size_t)lane * 16);

#pragma unroll
    for (int ks = 0; ks < 16; ++ks) {
        // All four B fragments up front: they are consumed interleaved below,
        // so all must stay live -> one 8-load clause, laddered waits.
        v16h b0 = *(const v16h*)&wbase[0 * 8192 + ks * 512];
        v16h b1 = *(const v16h*)&wbase[1 * 8192 + ks * 512];
        v16h b2 = *(const v16h*)&wbase[2 * 8192 + ks * 512];
        v16h b3 = *(const v16h*)&wbase[3 * 8192 + ks * 512];

        // Grouped by A fragment (m-outer): each a_m is loaded (cheap LDS)
        // and immediately used against b0..b3.
        {
            v16h a0 = *(const v16h*)&aswz[((0 * 16 + ks) * 32 + lane) * 16];
            acc[0][0] = __builtin_amdgcn_wmma_f32_16x16x32_f16(false, a0, false, b0, (short)0, acc[0][0], false, false);
            acc[1][0] = __builtin_amdgcn_wmma_f32_16x16x32_f16(false, a0, false, b1, (short)0, acc[1][0], false, false);
            acc[2][0] = __builtin_amdgcn_wmma_f32_16x16x32_f16(false, a0, false, b2, (short)0, acc[2][0], false, false);
            acc[3][0] = __builtin_amdgcn_wmma_f32_16x16x32_f16(false, a0, false, b3, (short)0, acc[3][0], false, false);
        }
        {
            v16h a1 = *(const v16h*)&aswz[((1 * 16 + ks) * 32 + lane) * 16];
            acc[0][1] = __builtin_amdgcn_wmma_f32_16x16x32_f16(false, a1, false, b0, (short)0, acc[0][1], false, false);
            acc[1][1] = __builtin_amdgcn_wmma_f32_16x16x32_f16(false, a1, false, b1, (short)0, acc[1][1], false, false);
            acc[2][1] = __builtin_amdgcn_wmma_f32_16x16x32_f16(false, a1, false, b2, (short)0, acc[2][1], false, false);
            acc[3][1] = __builtin_amdgcn_wmma_f32_16x16x32_f16(false, a1, false, b3, (short)0, acc[3][1], false, false);
        }
        {
            v16h a2 = *(const v16h*)&aswz[((2 * 16 + ks) * 32 + lane) * 16];
            acc[0][2] = __builtin_amdgcn_wmma_f32_16x16x32_f16(false, a2, false, b0, (short)0, acc[0][2], false, false);
            acc[1][2] = __builtin_amdgcn_wmma_f32_16x16x32_f16(false, a2, false, b1, (short)0, acc[1][2], false, false);
            acc[2][2] = __builtin_amdgcn_wmma_f32_16x16x32_f16(false, a2, false, b2, (short)0, acc[2][2], false, false);
            acc[3][2] = __builtin_amdgcn_wmma_f32_16x16x32_f16(false, a2, false, b3, (short)0, acc[3][2], false, false);
        }
        {
            v16h a3 = *(const v16h*)&aswz[((3 * 16 + ks) * 32 + lane) * 16];
            acc[0][3] = __builtin_amdgcn_wmma_f32_16x16x32_f16(false, a3, false, b0, (short)0, acc[0][3], false, false);
            acc[1][3] = __builtin_amdgcn_wmma_f32_16x16x32_f16(false, a3, false, b1, (short)0, acc[1][3], false, false);
            acc[2][3] = __builtin_amdgcn_wmma_f32_16x16x32_f16(false, a3, false, b2, (short)0, acc[2][3], false, false);
            acc[3][3] = __builtin_amdgcn_wmma_f32_16x16x32_f16(false, a3, false, b3, (short)0, acc[3][3], false, false);
        }
    }

    // Per-N-tile bias / context-vector values for this lane.
    float bnv[4], cnv[4];
#pragma unroll
    for (int i = 0; i < 4; ++i) {
        const int n = (wave * 4 + i) * 16 + (lane & 15);
        bnv[i] = bias[n];
        cnv[i] = cw[n];
    }

    // A fragments are dead; overlay the reduction buffer on the LDS stage.
    __syncthreads();
    float* sred = (float*)aswz;
    if (tid < MROWS) sred[tid] = 0.0f;
    __syncthreads();

    // Fuse tanh(pre+bias)*cw, reduce over N (16 lanes), then across waves.
#pragma unroll
    for (int m = 0; m < 4; ++m) {
        float s[8];
#pragma unroll
        for (int r = 0; r < 8; ++r) s[r] = 0.0f;
#pragma unroll
        for (int i = 0; i < 4; ++i)
#pragma unroll
            for (int r = 0; r < 8; ++r)
                s[r] += fast_tanh(acc[i][m][r] + bnv[i]) * cnv[i];
#pragma unroll
        for (int r = 0; r < 8; ++r) {
            float v = s[r];
            v += __shfl_xor(v, 1, 32);
            v += __shfl_xor(v, 2, 32);
            v += __shfl_xor(v, 4, 32);
            v += __shfl_xor(v, 8, 32);
            if ((lane & 15) == 0)      // C layout: M = r + 8*(lane>>4)
                atomicAdd(&sred[m * 16 + r + 8 * (lane >> 4)], v);
        }
    }
    __syncthreads();
    if (tid < MROWS)
        scores[(size_t)b * T_DIM + t0 + tid] = fast_tanh(sred[tid]);
}

// ---------------------------------------------------------------------------
// Kernel 2: per-batch softmax -> normalized attn weights (tiny: 512 KB I/O).
// ---------------------------------------------------------------------------
__global__ __launch_bounds__(512)
void attn_kernel(const float* __restrict__ scores, float* __restrict__ attn) {
    __shared__ float svals[T_DIM];   // 16 KB
    __shared__ float red[512];
    const int b   = blockIdx.x;
    const int tid = threadIdx.x;

    float m = -1e30f;
    for (int t = tid; t < T_DIM; t += 512) {
        float s = scores[(size_t)b * T_DIM + t];
        svals[t] = s;
        m = fmaxf(m, s);
    }
    red[tid] = m;
    __syncthreads();
    for (int s = 256; s > 0; s >>= 1) {
        if (tid < s) red[tid] = fmaxf(red[tid], red[tid + s]);
        __syncthreads();
    }
    const float maxv = red[0];
    __syncthreads();

    float sum = 0.f;
    for (int t = tid; t < T_DIM; t += 512) {
        float e = __expf(svals[t] - maxv);
        svals[t] = e;
        sum += e;
    }
    red[tid] = sum;
    __syncthreads();
    for (int s = 256; s > 0; s >>= 1) {
        if (tid < s) red[tid] += red[tid + s];
        __syncthreads();
    }
    const float inv = 1.0f / red[0];

    for (int t = tid; t < T_DIM; t += 512)
        attn[(size_t)b * T_DIM + t] = svals[t] * inv;
}

// ---------------------------------------------------------------------------
// Kernel 3: partial pooling. Grid = 32 batches x 16 T-chunks = 512 blocks so
// the 256 MiB embedding stream has enough memory-level parallelism for HBM.
// partial[(b*16 + c)*512 + d] = sum_{t in chunk c} attn[b,t] * emb[b,t,d]
// ---------------------------------------------------------------------------
__global__ __launch_bounds__(512)
void pool_partial_kernel(const float* __restrict__ emb, const float* __restrict__ attn,
                         float* __restrict__ partial) {
    __shared__ float sattn[CH_T];
    const int b   = blockIdx.x >> 4;
    const int c   = blockIdx.x & 15;
    const int tid = threadIdx.x;
    const int tbase = c * CH_T;

    if (tid < CH_T) sattn[tid] = attn[(size_t)b * T_DIM + tbase + tid];
    __syncthreads();

    const int d = tid;                 // 512 columns
    const float* ep = emb + ((size_t)b * T_DIM + tbase) * HID2 + d;
    float acc = 0.f;
    for (int t = 0; t < CH_T; ++t)
        acc = fmaf(sattn[t], ep[(size_t)t * HID2], acc);
    partial[((size_t)b * NCHUNK + c) * HID2 + d] = acc;
}

// ---------------------------------------------------------------------------
// Kernel 4: fold chunk partials -> pooled, then relu(pooled@w1+b1)@w2+b2.
// ---------------------------------------------------------------------------
__global__ __launch_bounds__(1024)
void cls_kernel(const float* __restrict__ partial, const float* __restrict__ w1,
                const float* __restrict__ b1, const float* __restrict__ w2,
                const float* __restrict__ b2, float* __restrict__ out) {
    __shared__ float sp[HID2];
    __shared__ float hid[H_CLS];
    __shared__ float red[1024];
    const int b   = blockIdx.x;
    const int tid = threadIdx.x;

    if (tid < HID2) {
        float s = 0.f;
#pragma unroll
        for (int c = 0; c < NCHUNK; ++c)
            s += partial[((size_t)b * NCHUNK + c) * HID2 + tid];
        sp[tid] = s;
    }
    __syncthreads();

    {   // hidden[j], j = tid
        float acc = b1[tid];
        for (int k = 0; k < HID2; ++k)
            acc = fmaf(sp[k], w1[k * H_CLS + tid], acc);
        hid[tid] = fmaxf(acc, 0.0f);
    }
    __syncthreads();

    const int o  = tid >> 7;     // 8 outputs x 128 reducers
    const int j0 = tid & 127;
    float p = 0.f;
    for (int j = j0; j < H_CLS; j += 128)
        p = fmaf(hid[j], w2[j * D_OUT + o], p);
    red[tid] = p;
    __syncthreads();
    for (int s = 64; s > 0; s >>= 1) {
        if (j0 < s) red[tid] += red[tid + s];
        __syncthreads();
    }
    if (j0 == 0) out[b * D_OUT + o] = red[tid] + b2[o];
}

// ---------------------------------------------------------------------------
extern "C" void kernel_launch(void* const* d_in, const int* in_sizes, int n_in,
                              void* d_out, int out_size, void* d_ws, size_t ws_size,
                              hipStream_t stream) {
    // setup_inputs order:
    // 0 numerical (unused by reference), 1 embedding, 2 weight, 3 bias,
    // 4 context_weight, 5 w1, 6 b1, 7 w2, 8 b2
    const float* emb  = (const float*)d_in[1];
    const float* w    = (const float*)d_in[2];
    const float* bias = (const float*)d_in[3];
    const float* cw   = (const float*)d_in[4];
    const float* w1   = (const float*)d_in[5];
    const float* b1   = (const float*)d_in[6];
    const float* w2   = (const float*)d_in[7];
    const float* b2   = (const float*)d_in[8];
    float* out        = (float*)d_out;

    // Workspace layout (2.5 MB total)
    char* ws = (char*)d_ws;
    _Float16* wf16    = (_Float16*)(ws);                       // 512 KB
    float*    scr     = (float*)(ws + (512 << 10));            // 512 KB
    float*    attn    = (float*)(ws + (1024 << 10));           // 512 KB
    float*    partial = (float*)(ws + (1536 << 10));           // 1 MB

    // 1) pre-swizzle weight to f16 B-fragments (re-run every call; deterministic)
    swizzle_weight_f16<<<(HID2 * HID2) / 256, 256, 0, stream>>>(w, wf16);

    // 2) WMMA GEMM + fused tanh/score reduction (64-row tiles)
    scores_kernel<<<B_DIM * (T_DIM / MROWS), 256, 0, stream>>>(emb, wf16, bias, cw, scr);

    // 3) softmax -> normalized attention weights
    attn_kernel<<<B_DIM, 512, 0, stream>>>(scr, attn);

    // 4) chunked attention pooling (HBM streaming pass, 512 blocks)
    pool_partial_kernel<<<B_DIM * NCHUNK, 512, 0, stream>>>(emb, attn, partial);

    // 5) classifier head (+ deterministic partial fold)
    cls_kernel<<<B_DIM, 1024, 0, stream>>>(partial, w1, b1, w2, b2, out);
}
